// N3Net_7335804142148
// MI455X (gfx1250) — compile-verified, matrix operand order checked by reading, not run on hardware
//
#include <hip/hip_runtime.h>
#include <hip/hip_bf16.h>

// ---------------------------------------------------------------------------
// N3Net (DnCNN + non-local patch block) for gfx1250.
// bf16 WMMA implicit-GEMM convs + WMMA patch-Gram, branch-free MMA loops.
// ---------------------------------------------------------------------------

typedef __attribute__((ext_vector_type(16))) __bf16 bf16x16;
typedef __attribute__((ext_vector_type(8)))  float  f32x8;

#define HH 160
#define WW 160
#define BB 8
#define FH 31          // (160-10)/5+1
#define NP (FH*FH)     // 961 patches
#define NPP 976        // padded to multiple of 16
#define PK 800         // 8 ch * 10*10

union FragAB { bf16x16 v; uint4 q[2]; unsigned u[8]; };

// bank-swizzled im2col LDS index: dword d (=k/2), pixel n (0..31)
#define SCI(d, n) ((d) * 32 + ((n) ^ ((((d) >> 2) & 1) << 4)))

// --------------------------- layer prep ------------------------------------
// wp: zero-padded bf16 weights [64][KP]; scale/shift fold bias + BN.
__global__ void __launch_bounds__(256)
prep_layer_kernel(const float* __restrict__ w, const float* __restrict__ b,
                  const float* __restrict__ g, const float* __restrict__ beta,
                  const float* __restrict__ m, const float* __restrict__ v,
                  __bf16* __restrict__ wp, float* __restrict__ scale,
                  float* __restrict__ shift, int K, int KP, int cout, int has_bn)
{
    int i = blockIdx.x * 256 + threadIdx.x;
    if (i < 64 * KP) {
        int mo = i / KP, k = i % KP;
        float vv = (mo < cout && k < K) ? w[mo * K + k] : 0.f;
        wp[i] = (__bf16)vv;
    }
    if (i < cout) {
        float s = 1.f, sh = b[i];
        if (has_bn) {
            s  = g[i] * rsqrtf(v[i] + 1e-5f);
            sh = b[i] * s + beta[i] - m[i] * s;
        }
        scale[i] = s; shift[i] = sh;
    }
}

// --------------------------- conv3x3 (implicit GEMM, WMMA bf16) ------------
// Block: 256 thr = 8 waves, tile 64(cout) x 32(pixels), one output row chunk.
// Stage 1: coalesced halo (3 x 34 x CINP ch) -> LDS (bf16 bits, zero padded).
// Stage 2: im2col repack -> sCol[k/2][32] dwords (2 bf16 per dword, swizzled).
// MMA loop: A = 2 x b128 global loads, B = 8 x ds_load_b32, branch-free.
template<int CIN>
__global__ void __launch_bounds__(256)
conv3x3_kernel(const float* __restrict__ in, const __bf16* __restrict__ wp,
               const float* __restrict__ scale, const float* __restrict__ shift,
               float* __restrict__ out, int cout, int relu)
{
    constexpr int K    = CIN * 9;
    constexpr int KP   = (K + 31) & ~31;     // 32 / 96 / 576
    constexpr int CINP = (KP + 8) / 9;       // staged channels (zero padded)

    __shared__ unsigned short sHalo[CINP * 3 * 34];
    __shared__ unsigned       sCol[(KP / 2) * 32];

    int chunk = blockIdx.x % 5;
    int h     = (blockIdx.x / 5) % HH;
    int b     = blockIdx.x / (5 * HH);
    int w0    = chunk * 32;
    int tid   = threadIdx.x;

    // ---- stage 1: halo load (coalesced, zero-padded channels/borders) ----
    for (int i = tid; i < CINP * 3 * 34; i += 256) {
        int col = i % 34;
        int rr  = (i / 34) % 3;
        int ci  = i / (34 * 3);
        int hh  = h - 1 + rr;
        int ww  = w0 - 1 + col;
        float vv = 0.f;
        if (ci < CIN && hh >= 0 && hh < HH && ww >= 0 && ww < WW)
            vv = in[((b * CIN + ci) * HH + hh) * WW + ww];
        sHalo[i] = __builtin_bit_cast(unsigned short, (__bf16)vv);
    }
    __syncthreads();

    // ---- stage 2: im2col repack into dword-packed K-major tile ----
    for (int i = tid; i < (KP / 2) * 32; i += 256) {
        int d = i >> 5, n = i & 31;
        int k0 = 2 * d, k1 = 2 * d + 1;
        int ci0 = k0 / 9, r0 = k0 % 9;
        int ci1 = k1 / 9, r1 = k1 % 9;
        unsigned lo = sHalo[(ci0 * 3 + r0 / 3) * 34 + n + r0 % 3];
        unsigned hi = sHalo[(ci1 * 3 + r1 / 3) * 34 + n + r1 % 3];
        sCol[SCI(d, n)] = lo | (hi << 16);
    }
    __syncthreads();

    // ---- stage 3: WMMA ----
    int wave = tid >> 5, lane = tid & 31;
    int mt = wave & 3, nt = wave >> 2;
    if (mt * 16 >= cout) return;          // wave-uniform, after all barriers

    int ml  = lane & 15;
    int hlf = lane >> 4;
    int n   = nt * 16 + ml;               // pixel within block
    int rowm = mt * 16 + ml;              // wp padded to 64 rows -> always valid
    const uint4* wrow4 = reinterpret_cast<const uint4*>(wp + rowm * KP);

    f32x8 acc = {0.f,0.f,0.f,0.f,0.f,0.f,0.f,0.f};
#pragma unroll
    for (int k0 = 0; k0 < KP; k0 += 32) {
        FragAB a, bb;
        int i0 = (k0 + hlf * 8) >> 3;     // uint4 index into weight row
        a.q[0] = wrow4[i0];
        a.q[1] = wrow4[i0 + 2];
        int d0 = (k0 + hlf * 8) >> 1;     // dword (k/2) index into sCol
#pragma unroll
        for (int j = 0; j < 4; ++j) {
            bb.u[j]     = sCol[SCI(d0 + j,     n)];
            bb.u[4 + j] = sCol[SCI(d0 + 8 + j, n)];
        }
        acc = __builtin_amdgcn_wmma_f32_16x16x32_bf16(
                  false, a.v, false, bb.v, (short)0, acc, false, false);
    }

    int wpix = w0 + n;                    // D: lane%16 = N(pixel)
#pragma unroll
    for (int vv = 0; vv < 8; ++vv) {      //    VGPR vv + 8*hlf = M(cout)
        int mo = mt * 16 + vv + 8 * hlf;
        if (mo < cout) {
            float r = acc[vv] * scale[mo] + shift[mo];
            if (relu) r = fmaxf(r, 0.f);
            out[((b * cout + mo) * HH + h) * WW + wpix] = r;
        }
    }
}

// --------------------------- n3block helpers -------------------------------
__device__ inline float patch_elem(const float* __restrict__ E, int b, int pn, int k)
{
    if (pn >= NP) return 0.f;
    int i = pn / FH, j = pn % FH;
    int c = k / 100, r = (k % 100) / 10, q = k % 10;
    return E[((b * 8 + c) * HH + (i * 5 + r)) * WW + (j * 5 + q)];
}

__global__ void __launch_bounds__(256)
patch_sq_kernel(const float* __restrict__ E, float* __restrict__ sq)
{
    int t = blockIdx.x * 256 + threadIdx.x;
    if (t >= BB * NP) return;
    int b = t / NP, n = t % NP;
    float s = 0.f;
    for (int k = 0; k < PK; ++k) { float v = patch_elem(E, b, n, k); s += v * v; }
    sq[t] = s;
}

// Pack patches: P[b][NPP][PK] bf16 (rows >= NP zero) -> contiguous GEMM rows.
__global__ void __launch_bounds__(256)
patch_pack_kernel(const float* __restrict__ E, __bf16* __restrict__ P)
{
    long t = (long)blockIdx.x * 256 + threadIdx.x;
    const long total = (long)BB * NPP * PK;
    if (t >= total) return;
    int k = (int)(t % PK);
    int n = (int)((t / PK) % NPP);
    int b = (int)(t / ((long)PK * NPP));
    P[t] = (__bf16)patch_elem(E, b, n, k);
}

// Gram via WMMA on packed rows: d2[b,row,col] = sq[row]+sq[col]-2<P_row,P_col>.
// grid.x = BB * 61 * 8 ; wave -> tm = grp*8 + wave.
__global__ void __launch_bounds__(256)
gram_kernel(const __bf16* __restrict__ P, const float* __restrict__ sq,
            float* __restrict__ D)
{
    int gid = blockIdx.x;
    int grp = gid % 8;
    int tn  = (gid / 8) % 61;
    int b   = gid / (8 * 61);

    int wave = threadIdx.x >> 5, lane = threadIdx.x & 31;
    int tm = grp * 8 + wave;
    if (tm >= 61) return;                 // wave-uniform

    int ml = lane & 15, hlf = lane >> 4;
    const uint4* arow = reinterpret_cast<const uint4*>(
        P + ((long)b * NPP + tn * 16 + ml) * PK);
    const uint4* brow = reinterpret_cast<const uint4*>(
        P + ((long)b * NPP + tm * 16 + ml) * PK);

    f32x8 acc = {0.f,0.f,0.f,0.f,0.f,0.f,0.f,0.f};
#pragma unroll 5
    for (int k0 = 0; k0 < PK; k0 += 32) {
        FragAB a, bb;
        int i0 = (k0 + hlf * 8) >> 3;
        a.q[0]  = arow[i0];     a.q[1]  = arow[i0 + 2];
        bb.q[0] = brow[i0];     bb.q[1] = brow[i0 + 2];
        acc = __builtin_amdgcn_wmma_f32_16x16x32_bf16(
                  false, a.v, false, bb.v, (short)0, acc, false, false);
    }

    int col = tm * 16 + ml;
    if (col < NP) {
        float sqc = sq[b * NP + col];
#pragma unroll
        for (int vv = 0; vv < 8; ++vv) {
            int row = tn * 16 + vv + 8 * hlf;
            if (row < NP) {
                float d2 = sq[b * NP + row] + sqc - 2.f * acc[vv];
                D[((long)b * NP + row) * NP + col] = d2;
            }
        }
    }
}

// top-8 smallest d2 per row, lax.top_k tie semantics (equal -> lower index).
__global__ void __launch_bounds__(256)
topk_kernel(const float* __restrict__ D, int* __restrict__ idx)
{
    int t = blockIdx.x * 256 + threadIdx.x;
    if (t >= BB * NP) return;
    const float* row = D + (long)t * NP;
    float lastv = -1e30f; int lasti = -1;
    for (int k = 0; k < 8; ++k) {
        float bestv = 1e30f; int besti = 0;
        for (int j = 0; j < NP; ++j) {
            float v = row[j];
            bool after = (v > lastv) || (v == lastv && j > lasti);
            if (after && v < bestv) { bestv = v; besti = j; }
        }
        idx[t * 8 + k] = besti;
        lastv = bestv; lasti = besti;
    }
}

// Overlap-add gather: out[b][kk*8+c][h][w] += E from self (kk=0) or idx[n][kk].
__global__ void __launch_bounds__(256)
n3gather_kernel(const float* __restrict__ E, const int* __restrict__ idx,
                float* __restrict__ out)
{
    long t = (long)blockIdx.x * 256 + threadIdx.x;
    const long total = (long)BB * 64 * HH * WW;
    if (t >= total) return;
    int w  = (int)(t % WW);
    int h  = (int)((t / WW) % HH);
    int ch = (int)((t / (WW * HH)) % 64);
    int b  = (int)(t / ((long)WW * HH * 64));
    int kk = ch >> 3, c = ch & 7;

    int imin = (h >= 9) ? (h - 5) / 5 : 0;
    int imax = min(FH - 1, h / 5);
    int jmin = (w >= 9) ? (w - 5) / 5 : 0;
    int jmax = min(FH - 1, w / 5);

    float acc = 0.f;
    for (int i = imin; i <= imax; ++i)
        for (int j = jmin; j <= jmax; ++j) {
            int n   = i * FH + j;
            int src = (kk == 0) ? n : idx[(b * NP + n) * 8 + kk];
            int si = src / FH, sj = src % FH;
            acc += E[((b * 8 + c) * HH + (si * 5 + h - i * 5)) * WW
                     + (sj * 5 + w - j * 5)];
        }
    out[t] = acc;
}

// --------------------------- host orchestration ----------------------------
extern "C" void kernel_launch(void* const* d_in, const int* in_sizes, int n_in,
                              void* d_out, int out_size, void* d_ws, size_t ws_size,
                              hipStream_t stream)
{
    (void)in_sizes; (void)n_in; (void)out_size; (void)ws_size;
    auto L = [&](int i) { return (const float*)d_in[i]; };

    char* ws = (char*)d_ws;
    size_t off = 0;
    auto alloc = [&](size_t bytes) {
        size_t o = off; off = (off + bytes + 255) & ~(size_t)255; return o;
    };
    const size_t ACTB = (size_t)BB * 64 * HH * WW * sizeof(float);
    float*  ACT0 = (float*)(ws + alloc(ACTB));
    float*  ACT1 = (float*)(ws + alloc(ACTB));
    float*  Y8   = (float*)(ws + alloc((size_t)BB * 8 * HH * WW * 4));
    float*  EB   = (float*)(ws + alloc((size_t)BB * 8 * HH * WW * 4));
    float*  SQ   = (float*)(ws + alloc((size_t)BB * NP * 4));
    float*  DM   = (float*)(ws + alloc((size_t)BB * NP * NP * 4));
    int*    IDX  = (int*)  (ws + alloc((size_t)BB * NP * 8 * 4));
    __bf16* WP   = (__bf16*)(ws + alloc((size_t)64 * 576 * 2));
    __bf16* PB   = (__bf16*)(ws + alloc((size_t)BB * NPP * PK * 2));
    float*  SC   = (float*)(ws + alloc(64 * 4));
    float*  SH   = (float*)(ws + alloc(64 * 4));

    auto conv = [&](const float* in, float* out, const float* w, const float* b,
                    const float* g, const float* be, const float* m, const float* v,
                    int cin, int cout, int relu) {
        int K = cin * 9, KP = (K + 31) & ~31;
        int tot = 64 * KP;
        prep_layer_kernel<<<(tot + 255) / 256, 256, 0, stream>>>(
            w, b, g, be, m, v, WP, SC, SH, K, KP, cout, g ? 1 : 0);
        dim3 grid(BB * HH * 5);
        if (cin == 3)
            conv3x3_kernel<3><<<grid, 256, 0, stream>>>(in, WP, SC, SH, out, cout, relu);
        else if (cin == 8)
            conv3x3_kernel<8><<<grid, 256, 0, stream>>>(in, WP, SC, SH, out, cout, relu);
        else
            conv3x3_kernel<64><<<grid, 256, 0, stream>>>(in, WP, SC, SH, out, cout, relu);
    };
    // cbr leaf order (sorted): b,+1 beta,+2 g,+3 m,+4 v,+5 w
    auto cbr = [&](const float* in, float* out, int base, int cin) {
        conv(in, out, L(base + 5), L(base + 0), L(base + 2), L(base + 1),
             L(base + 3), L(base + 4), cin, 64, 1);
    };
    // dncnn leaves: in.b,in.w, 4x cbr(6), out.b,out.w  (28 leaves)
    auto dncnn = [&](const float* in, float* out, int b0, int cin, int cout) {
        conv(in, ACT0, L(b0 + 1), L(b0 + 0), nullptr, nullptr, nullptr, nullptr,
             cin, 64, 1);
        cbr(ACT0, ACT1, b0 + 2,  64);
        cbr(ACT1, ACT0, b0 + 8,  64);
        cbr(ACT0, ACT1, b0 + 14, 64);
        cbr(ACT1, ACT0, b0 + 20, 64);
        conv(ACT0, out, L(b0 + 27), L(b0 + 26), nullptr, nullptr, nullptr, nullptr,
             64, cout, 0);
    };
    // n3 leaves: emb = [cbr(6), cbr(6), conv.b, conv.w]; temp branch is dead.
    auto n3 = [&](const float* y8, float* out64, int n0) {
        cbr(y8, ACT0, n0 + 0, 8);
        cbr(ACT0, ACT1, n0 + 6, 64);
        conv(ACT1, EB, L(n0 + 13), L(n0 + 12), nullptr, nullptr, nullptr, nullptr,
             64, 8, 0);
        patch_sq_kernel<<<(BB * NP + 255) / 256, 256, 0, stream>>>(EB, SQ);
        long ptot = (long)BB * NPP * PK;
        patch_pack_kernel<<<(int)((ptot + 255) / 256), 256, 0, stream>>>(EB, PB);
        gram_kernel<<<BB * 61 * 8, 256, 0, stream>>>(PB, SQ, DM);
        topk_kernel<<<(BB * NP + 255) / 256, 256, 0, stream>>>(DM, IDX);
        long tot = (long)BB * 64 * HH * WW;
        n3gather_kernel<<<(int)((tot + 255) / 256), 256, 0, stream>>>(EB, IDX, out64);
    };

    // model: d1 -> n1 -> d2 -> n2 -> recon   (leaf bases per sorted-key order)
    dncnn(L(0), Y8, 1, 3, 8);               // d1
    n3(Y8, ACT1, 57);                        // n1 -> 64ch in ACT1
    dncnn(ACT1, Y8, 29, 64, 8);              // d2
    n3(Y8, ACT1, 85);                        // n2
    dncnn(ACT1, (float*)d_out, 113, 64, 3);  // recon -> (8,3,160,160) fp32
}
